// TuckerLinearLayer_83726092468962
// MI455X (gfx1250) — matrix operand order repaired and checked by reading, not compiled, not run on hardware
//
#include <hip/hip_runtime.h>
#include <hip/hip_bf16.h>

// Tucker-factorized linear layer for MI455X (gfx1250, wave32, WMMA).
//
// y[n,abc] = sum_{def} W[abc,def] x[n,def],  W = core ×ijk f0,f1,f2 ×lmn f3,f4,f5
// Factorized: t1 = x @ (f3⊗f4⊗f5)  (K=4096 -> 512)
//             Wh = (f0⊗f1⊗f2) @ core^T   (4096 x 512)
//             y  = t1 @ Wh^T + bias
// ~36 GFLOP total vs 137 GFLOP dense; ~190 MB traffic -> memory-bound (~8us
// at 23.3 TB/s). bf16 WMMA (v_wmma_f32_16x16x32_bf16, f32 accum) keeps the
// GEMMs under the HBM roofline; fp32 WMMA (16x16x4) would be ~8x slower and
// compute-bound.

typedef __bf16 bf16_t;
typedef __attribute__((ext_vector_type(16))) __bf16 v16bf;
typedef __attribute__((ext_vector_type(8)))  __bf16 v8bf;
typedef __attribute__((ext_vector_type(8)))  float  v8f;

union ABFrag { v16bf v; v8bf h[2]; };

// A fragment (16x32 bf16, row-major source, row stride lda elements).
// Lane = 16*h + m holds row M=m, K = {8h..8h+7, 16+8h..16+8h+7}  (ISA 7.12.2).
__device__ __forceinline__ v16bf load_a_frag(const bf16_t* __restrict__ A,
                                             int lda, int row0, int k0, int lane) {
  const int m = lane & 15, h = lane >> 4;
  const bf16_t* p = A + (size_t)(row0 + m) * (size_t)lda + (k0 + 8 * h);
  ABFrag f;
  f.h[0] = *(const v8bf*)(p);        // K = 8h+0 .. 8h+7
  f.h[1] = *(const v8bf*)(p + 16);   // K = 16+8h+0 .. 16+8h+7
  return f.v;
}

// B fragment (32x16 bf16) from Bt stored [N][K] row-major (row stride ldb).
// Lane = 16*h + n holds column N=n, K = {8h..8h+7, 16+8h..16+8h+7} (matches A).
__device__ __forceinline__ v16bf load_b_frag(const bf16_t* __restrict__ Bt,
                                             int ldb, int col0, int k0, int lane) {
  const int n = lane & 15, h = lane >> 4;
  const bf16_t* p = Bt + (size_t)(col0 + n) * (size_t)ldb + (k0 + 8 * h);
  ABFrag f;
  f.h[0] = *(const v8bf*)(p);
  f.h[1] = *(const v8bf*)(p + 16);
  return f.v;
}

// D = A(MxK) . Bt(NxK)^T.  Block = 256 threads = 8 waves (4 in M, 2 in N).
// Block tile 128x64; each wave computes a 32x32 tile as 2x2 WMMA fragments.
// M % 128 == 0, N % 64 == 0, K % 32 == 0 (true for all three GEMMs here).
template <bool OUT_BF16, bool ADD_BIAS>
__global__ __launch_bounds__(256) void wmma_gemm_bf16(
    const bf16_t* __restrict__ A, const bf16_t* __restrict__ Bt,
    void* __restrict__ D, const float* __restrict__ bias,
    int K, int lda, int ldb, int ldd) {
  const int lane = threadIdx.x & 31;
  const int wave = threadIdx.x >> 5;
  const int row0 = blockIdx.y * 128 + (wave & 3) * 32;
  const int col0 = blockIdx.x * 64 + (wave >> 2) * 32;

  v8f acc00 = {}, acc01 = {}, acc10 = {}, acc11 = {};

  for (int k0 = 0; k0 < K; k0 += 32) {
    v16bf a0 = load_a_frag(A, lda, row0,      k0, lane);
    v16bf a1 = load_a_frag(A, lda, row0 + 16, k0, lane);
    v16bf b0 = load_b_frag(Bt, ldb, col0,      k0, lane);
    v16bf b1 = load_b_frag(Bt, ldb, col0 + 16, k0, lane);
    acc00 = __builtin_amdgcn_wmma_f32_16x16x32_bf16(false, a0, false, b0,
                                                    (short)0, acc00, false, false);
    acc01 = __builtin_amdgcn_wmma_f32_16x16x32_bf16(false, a0, false, b1,
                                                    (short)0, acc01, false, false);
    acc10 = __builtin_amdgcn_wmma_f32_16x16x32_bf16(false, a1, false, b0,
                                                    (short)0, acc10, false, false);
    acc11 = __builtin_amdgcn_wmma_f32_16x16x32_bf16(false, a1, false, b1,
                                                    (short)0, acc11, false, false);
  }

  // C/D layout: VGPR r -> (lanes 0-15: M=r, N=lane) (lanes 16-31: M=8+r, N=lane-16)
  const int n = lane & 15;
  const int g = lane >> 4;
  auto store_tile = [&](const v8f& d, int r0, int c0) {
    const int col = c0 + n;
    const float b = ADD_BIAS ? bias[col] : 0.0f;
#pragma unroll
    for (int r = 0; r < 8; ++r) {
      const int row = r0 + 8 * g + r;
      const float val = d[r] + b;
      if constexpr (OUT_BF16)
        ((bf16_t*)D)[(size_t)row * (size_t)ldd + col] = (bf16_t)val;
      else
        ((float*)D)[(size_t)row * (size_t)ldd + col] = val;
    }
  };
  store_tile(acc00, row0,      col0);
  store_tile(acc01, row0,      col0 + 16);
  store_tile(acc10, row0 + 16, col0);
  store_tile(acc11, row0 + 16, col0 + 16);
}

// ---- elementwise precompute kernels (tiny vs the GEMMs) ----

__global__ void convert_f32_to_bf16(const float* __restrict__ src,
                                    bf16_t* __restrict__ dst, int n4) {
  int i = (blockIdx.x * blockDim.x + threadIdx.x);
  if (i < n4) {
    float4 v = ((const float4*)src)[i];
    bf16_t* o = dst + 4 * (size_t)i;
    o[0] = (bf16_t)v.x; o[1] = (bf16_t)v.y; o[2] = (bf16_t)v.z; o[3] = (bf16_t)v.w;
  }
}

// MinT[(l,m,nn)][(d,e,f)] = f3[d,l]*f4[e,m]*f5[f,nn]   (512 x 4096)
__global__ void build_minT(const float* __restrict__ f3, const float* __restrict__ f4,
                           const float* __restrict__ f5, bf16_t* __restrict__ out) {
  int idx = blockIdx.x * blockDim.x + threadIdx.x;   // 512*4096 = 2M
  int col = idx & 4095;          // def
  int row = idx >> 12;           // lmn
  int d = col >> 8, e = (col >> 4) & 15, f = col & 15;
  int l = row >> 6, m = (row >> 3) & 7, nn = row & 7;
  out[idx] = (bf16_t)(f3[d * 8 + l] * f4[e * 8 + m] * f5[f * 8 + nn]);
}

// Mo[(a,b,c)][(i,j,k)] = f0[a,i]*f1[b,j]*f2[c,k]   (4096 x 512)
__global__ void build_mo(const float* __restrict__ f0, const float* __restrict__ f1,
                         const float* __restrict__ f2, bf16_t* __restrict__ out) {
  int idx = blockIdx.x * blockDim.x + threadIdx.x;   // 4096*512 = 2M
  int col = idx & 511;           // ijk
  int row = idx >> 9;            // abc
  int a = row >> 8, b = (row >> 4) & 15, c = row & 15;
  int i = col >> 6, j = (col >> 3) & 7, k = col & 7;
  out[idx] = (bf16_t)(f0[a * 8 + i] * f1[b * 8 + j] * f2[c * 8 + k]);
}

// CmatT[(l,m,n)][(i,j,k)] = core[i,j,k,l,m,n]   (512 x 512 transpose of core)
__global__ void build_cmatT(const float* __restrict__ core, bf16_t* __restrict__ out) {
  int idx = blockIdx.x * blockDim.x + threadIdx.x;   // 512*512 = 256K
  int row = idx >> 9;            // lmn
  int col = idx & 511;           // ijk
  out[idx] = (bf16_t)core[(size_t)col * 512 + row];
}

extern "C" void kernel_launch(void* const* d_in, const int* in_sizes, int n_in,
                              void* d_out, int out_size, void* d_ws, size_t ws_size,
                              hipStream_t stream) {
  const float* x    = (const float*)d_in[0];
  const float* core = (const float*)d_in[1];
  const float* f0   = (const float*)d_in[2];
  const float* f1   = (const float*)d_in[3];
  const float* f2   = (const float*)d_in[4];
  const float* f3   = (const float*)d_in[5];
  const float* f4   = (const float*)d_in[6];
  const float* f5   = (const float*)d_in[7];
  const float* bias = (const float*)d_in[8];

  const int IN = 4096, OUT = 4096, R = 512;        // prod(IN_F), prod(OUT_F), 8^3
  const int NB = in_sizes[0] / IN;                 // batch = 4096

  char* ws = (char*)d_ws;
  bf16_t* Xbf  = (bf16_t*)ws; ws += (size_t)NB * IN * 2;   // 32 MB
  bf16_t* MinT = (bf16_t*)ws; ws += (size_t)R * IN * 2;    //  4 MB
  bf16_t* Mo   = (bf16_t*)ws; ws += (size_t)OUT * R * 2;   //  4 MB
  bf16_t* CmT  = (bf16_t*)ws; ws += (size_t)R * R * 2;     // .5 MB
  bf16_t* Wh   = (bf16_t*)ws; ws += (size_t)OUT * R * 2;   //  4 MB
  bf16_t* T1   = (bf16_t*)ws; ws += (size_t)NB * R * 2;    //  4 MB

  // --- precompute (elementwise, ~20M elements total) ---
  {
    int n4 = (NB * IN) / 4;
    convert_f32_to_bf16<<<(n4 + 255) / 256, 256, 0, stream>>>(x, Xbf, n4);
  }
  build_minT<<<(R * IN) / 256, 256, 0, stream>>>(f3, f4, f5, MinT);
  build_mo<<<(OUT * R) / 256, 256, 0, stream>>>(f0, f1, f2, Mo);
  build_cmatT<<<(R * R) / 256, 256, 0, stream>>>(core, CmT);

  // --- G1: Wh[abc][lmn] = sum_ijk Mo[abc][ijk] * core[ijk][lmn]   (bf16 out) ---
  {
    dim3 grid(R / 64, OUT / 128);
    wmma_gemm_bf16<true, false><<<grid, 256, 0, stream>>>(
        Mo, CmT, Wh, nullptr, /*K=*/R, /*lda=*/R, /*ldb=*/R, /*ldd=*/R);
  }
  // --- G2: T1[n][lmn] = sum_def Xbf[n][def] * MinT[lmn][def]   (bf16 out) ---
  {
    dim3 grid(R / 64, NB / 128);
    wmma_gemm_bf16<true, false><<<grid, 256, 0, stream>>>(
        Xbf, MinT, T1, nullptr, /*K=*/IN, /*lda=*/IN, /*ldb=*/IN, /*ldd=*/R);
  }
  // --- G3: Y[n][abc] = sum_lmn T1[n][lmn] * Wh[abc][lmn] + bias   (f32 out) ---
  {
    dim3 grid(OUT / 64, NB / 128);
    wmma_gemm_bf16<false, true><<<grid, 256, 0, stream>>>(
        T1, Wh, d_out, bias, /*K=*/R, /*lda=*/R, /*ldb=*/R, /*ldd=*/OUT);
  }
}
